// MultiHeadAttention_2946347565519
// MI455X (gfx1250) — matrix operand order repaired and checked by reading, not compiled, not run on hardware
//
#include <hip/hip_runtime.h>
#include <math.h>

// ---------- types ----------
typedef __attribute__((ext_vector_type(16))) __bf16 v16bf;
typedef __attribute__((ext_vector_type(8)))  float  v8f;
typedef __attribute__((ext_vector_type(4)))  unsigned int   u32x4;
typedef __attribute__((ext_vector_type(4)))  float          f32x4;
typedef __attribute__((ext_vector_type(4)))  unsigned short u16x4;

#define E_DIM 1024
#define SEQ   2048
#define BATCH 2
#define HEADS 16
#define HDIM  64
#define MROWS 4096   // BATCH*SEQ

union FragBF {
    v16bf v;
    u32x4 q[2];
    unsigned short us[16];
};

__device__ __forceinline__ unsigned short f2bf(float f) {
    unsigned int u = __builtin_bit_cast(unsigned int, f);
    u += 0x7fffu + ((u >> 16) & 1u);          // round-to-nearest-even
    return (unsigned short)(u >> 16);
}

// ---------- f32 -> bf16 (vectorized x4) ----------
__global__ void cvt_f32_bf16(const float* __restrict__ in,
                             unsigned short* __restrict__ out, int n4) {
    int i = blockIdx.x * blockDim.x + threadIdx.x;
    if (i < n4) {
        f32x4 f = ((const f32x4*)in)[i];
        u16x4 o;
        o.x = f2bf(f.x); o.y = f2bf(f.y); o.z = f2bf(f.z); o.w = f2bf(f.w);
        ((u16x4*)out)[i] = o;
    }
}

// ---------- weight transpose + convert: W[k][n] f32 -> WT[n][k] bf16 ----------
__global__ void cvt_transpose_w(const float* __restrict__ W,
                                unsigned short* __restrict__ WT) {
    int n = blockIdx.x * 16 + threadIdx.x;
    int k = blockIdx.y * 16 + threadIdx.y;
    WT[(size_t)n * E_DIM + k] = f2bf(W[(size_t)k * E_DIM + n]);
}

// ---------- GEMM: C[m][n] = sum_k A[m][k]*BT[n][k] + bias[n] ----------
// Wave tile: 32(M) x 64(N)  -> 8 WMMAs per 12 b128 loads per k-step.
// Workgroup: 8 waves as 2M x 4N -> 64 x 256 block tile.
// mode 0: bf16 row-major out; mode 1: bf16 V-head-transposed out; mode 2: f32 out
__global__ __launch_bounds__(256) void gemm_proj(
    const unsigned short* __restrict__ A, const unsigned short* __restrict__ BT,
    const float* __restrict__ bias, unsigned short* __restrict__ Cbf,
    float* __restrict__ Cf, int mode) {
    const int lane = threadIdx.x & 31;
    const int wave = threadIdx.x >> 5;
    const int half = lane >> 4, l16 = lane & 15;
    const int row0 = blockIdx.y * 64 + (wave & 1) * 32;
    const int col0 = blockIdx.x * 256 + (wave >> 1) * 64;

    const v8f zero = {0.f, 0.f, 0.f, 0.f, 0.f, 0.f, 0.f, 0.f};
    v8f acc[2][4] = {{zero, zero, zero, zero}, {zero, zero, zero, zero}};

    const unsigned short* arow0 = A + (size_t)(row0 + l16) * E_DIM;
    const unsigned short* arow1 = A + (size_t)(row0 + 16 + l16) * E_DIM;
    for (int k0 = 0; k0 < E_DIM; k0 += 32) {
        FragBF a0, a1;
        {
            const u32x4* ap0 = (const u32x4*)(arow0 + k0 + 8 * half);
            a0.q[0] = ap0[0];
            a0.q[1] = ap0[2];
            const u32x4* ap1 = (const u32x4*)(arow1 + k0 + 8 * half);
            a1.q[0] = ap1[0];
            a1.q[1] = ap1[2];
        }
        if (k0 + 32 < E_DIM) {
            __builtin_prefetch(arow0 + k0 + 32 + 8 * half, 0, 0);
            __builtin_prefetch(arow1 + k0 + 32 + 8 * half, 0, 0);
        }
#pragma unroll
        for (int t = 0; t < 4; ++t) {
            FragBF b;
            const u32x4* bp =
                (const u32x4*)(BT + (size_t)(col0 + 16 * t + l16) * E_DIM + k0 + 8 * half);
            b.q[0] = bp[0];
            b.q[1] = bp[2];
            acc[0][t] = __builtin_amdgcn_wmma_f32_16x16x32_bf16(
                false, a0.v, false, b.v, (short)0, acc[0][t], false, false);
            acc[1][t] = __builtin_amdgcn_wmma_f32_16x16x32_bf16(
                false, a1.v, false, b.v, (short)0, acc[1][t], false, false);
        }
    }

#pragma unroll
    for (int u = 0; u < 2; ++u) {
#pragma unroll
        for (int t = 0; t < 4; ++t) {
            const int col = col0 + 16 * t + l16;
            const float bv = bias[col];
#pragma unroll
            for (int r = 0; r < 8; ++r) {
                const int row = row0 + u * 16 + r + 8 * half;
                const float o = acc[u][t][r] + bv;
                if (mode == 0) {
                    Cbf[(size_t)row * E_DIM + col] = f2bf(o);
                } else if (mode == 1) {
                    // Vt[(b*H + h)*64 + d][s]
                    const int b = row >> 11, s = row & 2047;
                    const int h = col >> 6, d = col & 63;
                    Cbf[((size_t)((b * HEADS + h) * HDIM + d)) * SEQ + s] = f2bf(o);
                } else {
                    Cf[(size_t)row * E_DIM + col] = o;
                }
            }
        }
    }
}

// ---------- flash attention: per wave 16 query rows of one (b,h) ----------
__global__ __launch_bounds__(128) void flash_attn(
    const unsigned short* __restrict__ Q, const unsigned short* __restrict__ K,
    const unsigned short* __restrict__ Vt, unsigned short* __restrict__ AO) {
    const int bh = blockIdx.y;
    const int b = bh >> 4, h = bh & 15;
    const int wave = threadIdx.x >> 5;
    const int lane = threadIdx.x & 31;
    const int half = lane >> 4, l16 = lane & 15;
    const int q0 = blockIdx.x * 64 + wave * 16;

    __shared__ unsigned short plds[4][16 * 32];
    unsigned short* myp = plds[wave];

    // Q A-fragments for d = 0..63 (two k-steps of 32), kept resident
    const unsigned short* qrow = Q + (size_t)(b * SEQ + q0 + l16) * E_DIM + h * HDIM;
    FragBF aq0, aq1;
    {
        const u32x4* p0 = (const u32x4*)(qrow + 8 * half);
        aq0.q[0] = p0[0]; aq0.q[1] = p0[2];
        const u32x4* p1 = (const u32x4*)(qrow + 32 + 8 * half);
        aq1.q[0] = p1[0]; aq1.q[1] = p1[2];
    }

    const v8f zero = {0.f, 0.f, 0.f, 0.f, 0.f, 0.f, 0.f, 0.f};
    v8f acc[4] = {zero, zero, zero, zero};
    float m_row[8], l_row[8];
#pragma unroll
    for (int r = 0; r < 8; ++r) { m_row[r] = -INFINITY; l_row[r] = 0.f; }

    const float scale = 0.125f;  // 1/sqrt(64)

    for (int kb = 0; kb < SEQ; kb += 32) {
        // ---- scores: 16 q-rows x 32 keys ----
        v8f st[2];
#pragma unroll
        for (int nt = 0; nt < 2; ++nt) {
            const unsigned short* krow =
                K + (size_t)(b * SEQ + kb + nt * 16 + l16) * E_DIM + h * HDIM;
            FragBF bk0, bk1;
            const u32x4* p0 = (const u32x4*)(krow + 8 * half);
            bk0.q[0] = p0[0]; bk0.q[1] = p0[2];
            const u32x4* p1 = (const u32x4*)(krow + 32 + 8 * half);
            bk1.q[0] = p1[0]; bk1.q[1] = p1[2];
            v8f s = __builtin_amdgcn_wmma_f32_16x16x32_bf16(
                false, aq0.v, false, bk0.v, (short)0, zero, false, false);
            s = __builtin_amdgcn_wmma_f32_16x16x32_bf16(
                false, aq1.v, false, bk1.v, (short)0, s, false, false);
#pragma unroll
            for (int r = 0; r < 8; ++r) s[r] *= scale;
            st[nt] = s;
        }

        // ---- online softmax (row stats replicated across each 16-lane half) ----
        float alpha[8];
#pragma unroll
        for (int r = 0; r < 8; ++r) {
            float mx = fmaxf(st[0][r], st[1][r]);
#pragma unroll
            for (int msk = 8; msk >= 1; msk >>= 1)
                mx = fmaxf(mx, __shfl_xor(mx, msk));
            const float mnew = fmaxf(m_row[r], mx);
            alpha[r] = __expf(m_row[r] - mnew);
            m_row[r] = mnew;
            const float p0 = __expf(st[0][r] - mnew);
            const float p1 = __expf(st[1][r] - mnew);
            st[0][r] = p0; st[1][r] = p1;
            float rs = p0 + p1;
#pragma unroll
            for (int msk = 8; msk >= 1; msk >>= 1)
                rs += __shfl_xor(rs, msk);
            l_row[r] = l_row[r] * alpha[r] + rs;
        }
#pragma unroll
        for (int t = 0; t < 4; ++t)
#pragma unroll
            for (int r = 0; r < 8; ++r) acc[t][r] *= alpha[r];

        // ---- re-layout P (C/D layout) -> A-fragment via LDS ----
#pragma unroll
        for (int r = 0; r < 8; ++r) {
            const int rr = r + 8 * half;
            myp[rr * 32 + l16]      = f2bf(st[0][r]);
            myp[rr * 32 + 16 + l16] = f2bf(st[1][r]);
        }
        __syncthreads();
        FragBF pa;
        pa.q[0] = *(const u32x4*)&myp[l16 * 32 + 8 * half];
        pa.q[1] = *(const u32x4*)&myp[l16 * 32 + 16 + 8 * half];
        __syncthreads();

        // ---- O += P @ V  (Vt[d][s] is contiguous in keys) ----
#pragma unroll
        for (int t = 0; t < 4; ++t) {
            const unsigned short* vrow =
                Vt + (size_t)(bh * HDIM + 16 * t + l16) * SEQ + kb + 8 * half;
            FragBF bv;
            const u32x4* vp = (const u32x4*)vrow;
            bv.q[0] = vp[0]; bv.q[1] = vp[2];
            acc[t] = __builtin_amdgcn_wmma_f32_16x16x32_bf16(
                false, pa.v, false, bv.v, (short)0, acc[t], false, false);
        }
    }

    // ---- normalize and write head output (bf16, [B*S, E] head-major cols) ----
#pragma unroll
    for (int t = 0; t < 4; ++t) {
        const int col = h * HDIM + 16 * t + l16;
#pragma unroll
        for (int r = 0; r < 8; ++r) {
            const int token = b * SEQ + q0 + r + 8 * half;
            AO[(size_t)token * E_DIM + col] = f2bf(acc[t][r] / l_row[r]);
        }
    }
}

// ---------- host launch ----------
extern "C" void kernel_launch(void* const* d_in, const int* in_sizes, int n_in,
                              void* d_out, int out_size, void* d_ws, size_t ws_size,
                              hipStream_t stream) {
    const float* q  = (const float*)d_in[0];
    const float* k  = (const float*)d_in[1];
    const float* v  = (const float*)d_in[2];
    const float* Wq = (const float*)d_in[3];
    const float* bq = (const float*)d_in[4];
    const float* Wk = (const float*)d_in[5];
    const float* bk = (const float*)d_in[6];
    const float* Wv = (const float*)d_in[7];
    const float* bv = (const float*)d_in[8];
    const float* Wo = (const float*)d_in[9];
    const float* bo = (const float*)d_in[10];
    float* out = (float*)d_out;

    unsigned short* ws = (unsigned short*)d_ws;
    const size_t M1 = 1u << 20;  // 1M ushorts = 2MB
    unsigned short* XQ  = ws;            // 4M each for [4096,1024] bf16
    unsigned short* XK  = ws + 4  * M1;
    unsigned short* XV  = ws + 8  * M1;
    unsigned short* QB  = ws + 12 * M1;
    unsigned short* KB  = ws + 16 * M1;
    unsigned short* VT  = ws + 20 * M1;  // [2048, 2048]
    unsigned short* AO  = ws + 24 * M1;
    unsigned short* WQT = ws + 28 * M1;  // 1M each
    unsigned short* WKT = ws + 29 * M1;
    unsigned short* WVT = ws + 30 * M1;
    unsigned short* WOT = ws + 31 * M1;

    const int n4 = (MROWS * E_DIM) / 4;  // 1,048,576
    cvt_f32_bf16<<<n4 / 256, 256, 0, stream>>>(q, XQ, n4);
    cvt_f32_bf16<<<n4 / 256, 256, 0, stream>>>(k, XK, n4);
    cvt_f32_bf16<<<n4 / 256, 256, 0, stream>>>(v, XV, n4);

    dim3 tb(16, 16), tg(64, 64);
    cvt_transpose_w<<<tg, tb, 0, stream>>>(Wq, WQT);
    cvt_transpose_w<<<tg, tb, 0, stream>>>(Wk, WKT);
    cvt_transpose_w<<<tg, tb, 0, stream>>>(Wv, WVT);
    cvt_transpose_w<<<tg, tb, 0, stream>>>(Wo, WOT);

    dim3 gg(4, 64);  // N/256, M/64
    gemm_proj<<<gg, 256, 0, stream>>>(XQ, WQT, bq, QB, nullptr, 0);
    gemm_proj<<<gg, 256, 0, stream>>>(XK, WKT, bk, KB, nullptr, 0);
    gemm_proj<<<gg, 256, 0, stream>>>(XV, WVT, bv, VT, nullptr, 1);

    flash_attn<<<dim3(SEQ / 64, BATCH * HEADS), 128, 0, stream>>>(QB, KB, VT, AO);

    gemm_proj<<<gg, 256, 0, stream>>>(AO, WOT, bo, nullptr, out, 2);
}
